// AttEncoder_37795712205376
// MI455X (gfx1250) — matrix operand order repaired and checked by reading, not compiled, not run on hardware
//
#include <hip/hip_runtime.h>
#include <hip/hip_bf16.h>

// ---------------------------------------------------------------------------
// Transformer encoder forward for MI455X (gfx1250), bf16 WMMA + fp32 accum.
// Weights are pre-transposed to [N][K] bf16 so all LDS staging is contiguous
// b128 traffic; GEMM uses async global->LDS double buffering when available.
// ---------------------------------------------------------------------------

typedef __attribute__((ext_vector_type(16))) __bf16 v16bf;
typedef __attribute__((ext_vector_type(8)))  float  v8f;
typedef int v4i_b128 __attribute__((vector_size(16)));   // matches async-LDS builtin param

#define NLAYER 6
#define NHEAD  8
#define DMODEL 512
#define DHEAD  64
#define DFFN   2048
#define NBATCH 8
#define SEQ    1024
#define BSROWS (NBATCH * SEQ)   // 8192

#if __has_builtin(__builtin_amdgcn_global_load_async_to_lds_b128) && \
    __has_builtin(__builtin_amdgcn_s_wait_asynccnt)
#define HAVE_ASYNC_LDS 1
#else
#define HAVE_ASYNC_LDS 0
#endif

// round-to-nearest-even f32 -> bf16 (raw u16)
__device__ __forceinline__ unsigned short f2bf(float f) {
  unsigned int u = __float_as_uint(f);
  unsigned int r = u + 0x7FFFu + ((u >> 16) & 1u);
  return (unsigned short)(r >> 16);
}

union FragBF { v16bf v; uint4 q[2]; };

// Load a 16-lane-striped bf16 WMMA fragment from LDS: two 16B chunks,
// second chunk `step2` bf16 elements after the first (16 for both A and B).
__device__ __forceinline__ v16bf load_frag(const unsigned short* p, int step2) {
  FragBF f;
  f.q[0] = *(const uint4*)(p);
  f.q[1] = *(const uint4*)(p + step2);
  return f.v;
}

// ---------------------------------------------------------------------------
// Generic bf16 GEMM:  C[M,N] = A[M,K] * Bt[N,K]^T (+bias) (+relu)
// A is [M,K] row-major bf16; B is pre-transposed [N,K] row-major bf16.
// Block tile 128x128, K-slab 64, 256 threads (8 waves), wave tile 32x64.
// Async LDS double-buffer when available, else register double-buffer.
// OUTVT: scatter-store bf16 output as V-transposed [B,H,DK,S] for attention.
// ---------------------------------------------------------------------------
#define GBM 128
#define GBN 128
#define LDS_S 72   // LDS row stride (bf16 elems); 144B keeps 16B alignment

template<bool HB, bool RELU, bool OUTF, bool OUTB, bool OUTVT>
__global__ __launch_bounds__(256) void gemm_bf16_wmma(
    const unsigned short* __restrict__ A,
    const unsigned short* __restrict__ B,
    const float* __restrict__ bias,
    float* __restrict__ Cf,
    unsigned short* __restrict__ Cb,
    int N, int K)
{
  const int tid  = threadIdx.x;
  const int lane = tid & 31;
  const int wave = tid >> 5;
  const int bm   = blockIdx.y * GBM;
  const int bn   = blockIdx.x * GBN;
  const int wm   = (wave >> 1) * 32;   // wave row offset inside block tile
  const int wn   = (wave & 1) * 64;    // wave col offset
  const int l15  = lane & 15;
  const int hi   = lane >> 4;

  v8f zero = {};
  v8f acc[2][4];
  #pragma unroll
  for (int i = 0; i < 2; i++)
    #pragma unroll
    for (int j = 0; j < 4; j++) acc[i][j] = zero;

  const int ksteps = K >> 6;

  // one compute pass over a staged 64-deep slab
  auto compute = [&](const unsigned short* pA, const unsigned short* pB) {
    #pragma unroll
    for (int ks = 0; ks < 2; ks++) {
      v16bf af[2], bfb[4];
      #pragma unroll
      for (int i = 0; i < 2; i++)
        af[i] = load_frag(pA + (wm + i * 16 + l15) * LDS_S + ks * 32 + hi * 8, 16);
      #pragma unroll
      for (int j = 0; j < 4; j++)
        bfb[j] = load_frag(pB + (wn + j * 16 + l15) * LDS_S + ks * 32 + hi * 8, 16);
      #pragma unroll
      for (int i = 0; i < 2; i++)
        #pragma unroll
        for (int j = 0; j < 4; j++)
          acc[i][j] = __builtin_amdgcn_wmma_f32_16x16x32_bf16(
              false, af[i], false, bfb[j], (short)0, acc[i][j], false, false);
    }
  };

#if HAVE_ASYNC_LDS
  // ---- async global->LDS double buffering (ASYNCcnt path) ----
  __shared__ __align__(16) unsigned short sA[2][GBM * LDS_S];
  __shared__ __align__(16) unsigned short sB[2][GBN * LDS_S];

  auto issue = [&](int buf, int k0) {
    #pragma unroll
    for (int c = 0; c < 4; c++) {
      int ch = c * 256 + tid;                  // 1024 16B chunks each
      int row = ch >> 3, kc = (ch & 7) << 3;   // 8 chunks per 64-deep row
      __builtin_amdgcn_global_load_async_to_lds_b128(
          (__attribute__((address_space(1))) v4i_b128*)(A + (size_t)(bm + row) * K + k0 + kc),
          (__attribute__((address_space(3))) v4i_b128*)(&sA[buf][row * LDS_S + kc]), 0, 0);
      __builtin_amdgcn_global_load_async_to_lds_b128(
          (__attribute__((address_space(1))) v4i_b128*)(B + (size_t)(bn + row) * K + k0 + kc),
          (__attribute__((address_space(3))) v4i_b128*)(&sB[buf][row * LDS_S + kc]), 0, 0);
    }
  };

  issue(0, 0);
  int buf = 0;
  for (int kt = 0; kt < ksteps; kt++) {
    __builtin_amdgcn_s_wait_asynccnt(0);   // slab kt resident in LDS
    __syncthreads();                       // all waves done with other buffer
    if (kt + 1 < ksteps) issue(buf ^ 1, (kt + 1) << 6);
    compute(sA[buf], sB[buf]);
    buf ^= 1;
  }
#else
  // ---- register double buffering fallback ----
  __shared__ __align__(16) unsigned short sA[GBM * LDS_S];
  __shared__ __align__(16) unsigned short sB[GBN * LDS_S];
  uint4 ra[4], rb[4];

  auto fetch = [&](int k0) {
    #pragma unroll
    for (int c = 0; c < 4; c++) {
      int ch = c * 256 + tid;
      int row = ch >> 3, kc = (ch & 7) << 3;
      ra[c] = *(const uint4*)(A + (size_t)(bm + row) * K + k0 + kc);
      rb[c] = *(const uint4*)(B + (size_t)(bn + row) * K + k0 + kc);
    }
  };

  fetch(0);
  for (int kt = 0; kt < ksteps; kt++) {
    #pragma unroll
    for (int c = 0; c < 4; c++) {
      int ch = c * 256 + tid;
      int row = ch >> 3, kc = (ch & 7) << 3;
      *(uint4*)(sA + row * LDS_S + kc) = ra[c];
      *(uint4*)(sB + row * LDS_S + kc) = rb[c];
    }
    __syncthreads();
    if (kt + 1 < ksteps) fetch((kt + 1) << 6);
    compute(sA, sB);
    __syncthreads();
  }
#endif

  // ---- epilogue: C/D layout: elem e -> M = e + 8*(lane>=16), N = lane&15 ----
  float bv[4];
  if (HB) {
    #pragma unroll
    for (int j = 0; j < 4; j++) bv[j] = bias[bn + wn + j * 16 + l15];
  }
  #pragma unroll
  for (int i = 0; i < 2; i++)
    #pragma unroll
    for (int j = 0; j < 4; j++)
      #pragma unroll
      for (int e = 0; e < 8; e++) {
        int gm = bm + wm + i * 16 + e + (hi << 3);
        int gn = bn + wn + j * 16 + l15;
        float v = acc[i][j][e];
        if (HB)   v += bv[j];
        if (RELU) v = fmaxf(v, 0.0f);
        if (OUTF) Cf[(size_t)gm * N + gn] = v;
        if (OUTB) Cb[(size_t)gm * N + gn] = f2bf(v);
        if (OUTVT) {
          // V output transposed for attention: [B, H, DK, S]
          int bb = gm >> 10, ss = gm & (SEQ - 1);
          int hh = gn >> 6,  dk = gn & (DHEAD - 1);
          Cb[(size_t)(((bb << 3) + hh) * DHEAD + dk) * SEQ + ss] = f2bf(v);
        }
      }
}

// ---------------------------------------------------------------------------
// Flash attention: one block per (b, h, 64-query tile). Q/K packed bf16
// [BS, H*DK]; V pre-transposed bf16 [B,H,DK,S]. Streams 16 key tiles of 64,
// online softmax, output bf16 packed [BS, H*DK].
// ---------------------------------------------------------------------------
#define TAS 72   // LDS stride (bf16) for 64-wide tiles; 144B, 16B aligned

__global__ __launch_bounds__(256) void attn_wmma(
    const unsigned short* __restrict__ Qp,
    const unsigned short* __restrict__ Kp,
    const unsigned short* __restrict__ Vt,
    unsigned short* __restrict__ Op,
    float scale)
{
  __shared__ __align__(16) unsigned short sQ[64 * TAS];
  __shared__ __align__(16) unsigned short sK[64 * TAS];
  __shared__ __align__(16) unsigned short sVt[64 * TAS]; // [dk][key]
  __shared__ __align__(16) unsigned short sP[64 * TAS];
  __shared__ float sS[64 * 66];
  __shared__ float sM[64], sL[64], sR[64];
  __shared__ float sRed[64 * 4];

  const int tid  = threadIdx.x;
  const int lane = tid & 31;
  const int wave = tid >> 5;
  const int l15  = lane & 15;
  const int hi   = lane >> 4;
  const int b    = blockIdx.z, h = blockIdx.y;
  const int q0   = blockIdx.x * 64;
  const int m0   = (wave >> 1) * 16;   // wave's query sub-tile
  const int wn   = (wave & 1) * 32;    // wave's col offset (keys / dk)
  const int srow = tid & 63;
  const int part = tid >> 6;
  const size_t vbase = (size_t)(b * NHEAD + h) * DHEAD * SEQ;

  if (tid < 64) { sM[tid] = -3.0e38f; sL[tid] = 0.0f; }

  // stage Q tile [64 x 64]
  #pragma unroll
  for (int c = 0; c < 2; c++) {
    int ch = tid * 2 + c;
    int row = ch >> 3, kc = (ch & 7) << 3;
    uint4 d = *(const uint4*)(Qp + (size_t)(b * SEQ + q0 + row) * DMODEL + h * DHEAD + kc);
    *(uint4*)(sQ + row * TAS + kc) = d;
  }
  __syncthreads();

  v8f acc[2]; acc[0] = (v8f){}; acc[1] = (v8f){};

  for (int kt = 0; kt < SEQ / 64; kt++) {
    // ---- stage K tile [key][dk] and V-transposed tile [dk][key] ----
    #pragma unroll
    for (int c = 0; c < 2; c++) {
      int ch = tid * 2 + c;
      int row = ch >> 3, kc = (ch & 7) << 3;
      uint4 dk_ = *(const uint4*)(Kp + (size_t)(b * SEQ + kt * 64 + row) * DMODEL + h * DHEAD + kc);
      *(uint4*)(sK + row * TAS + kc) = dk_;
      uint4 dv = *(const uint4*)(Vt + vbase + (size_t)row * SEQ + kt * 64 + kc);
      *(uint4*)(sVt + row * TAS + kc) = dv;
    }
    __syncthreads();

    // ---- scores: S = scale * Q @ K^T (accumulate over dk, 2 k-steps) ----
    {
      v8f sc[2]; sc[0] = (v8f){}; sc[1] = (v8f){};
      #pragma unroll
      for (int ks = 0; ks < 2; ks++) {
        v16bf aq = load_frag(sQ + (m0 + l15) * TAS + ks * 32 + hi * 8, 16);
        #pragma unroll
        for (int j = 0; j < 2; j++) {
          v16bf bk_ = load_frag(sK + (wn + j * 16 + l15) * TAS + ks * 32 + hi * 8, 16);
          sc[j] = __builtin_amdgcn_wmma_f32_16x16x32_bf16(
              false, aq, false, bk_, (short)0, sc[j], false, false);
        }
      }
      #pragma unroll
      for (int j = 0; j < 2; j++)
        #pragma unroll
        for (int e = 0; e < 8; e++)
          sS[(m0 + e + (hi << 3)) * 66 + wn + j * 16 + l15] = sc[j][e] * scale;
    }
    __syncthreads();

    // ---- online softmax: 4 threads per row, 16 cols each ----
    {
      float mx = -3.0e38f;
      #pragma unroll
      for (int c = 0; c < 16; c++) mx = fmaxf(mx, sS[srow * 66 + part * 16 + c]);
      sRed[srow * 4 + part] = mx;
    }
    __syncthreads();
    float mold = sM[srow];
    float tmax = fmaxf(fmaxf(sRed[srow * 4 + 0], sRed[srow * 4 + 1]),
                       fmaxf(sRed[srow * 4 + 2], sRed[srow * 4 + 3]));
    float mnew = fmaxf(mold, tmax);
    float ps = 0.0f;
    #pragma unroll
    for (int c = 0; c < 16; c++) {
      float p = __expf(sS[srow * 66 + part * 16 + c] - mnew);
      sP[srow * TAS + part * 16 + c] = f2bf(p);
      ps += p;
    }
    __syncthreads();
    sRed[srow * 4 + part] = ps;
    if (part == 0) { sR[srow] = __expf(mold - mnew); sM[srow] = mnew; }
    __syncthreads();
    if (part == 0) {
      float ts = sRed[srow * 4 + 0] + sRed[srow * 4 + 1] +
                 sRed[srow * 4 + 2] + sRed[srow * 4 + 3];
      sL[srow] = sL[srow] * sR[srow] + ts;
    }
    __syncthreads();

    // ---- rescale accumulators, then O += P @ V ----
    #pragma unroll
    for (int j = 0; j < 2; j++)
      #pragma unroll
      for (int e = 0; e < 8; e++)
        acc[j][e] *= sR[m0 + e + (hi << 3)];
    #pragma unroll
    for (int ks = 0; ks < 2; ks++) {
      v16bf ap = load_frag(sP + (m0 + l15) * TAS + ks * 32 + hi * 8, 16);
      #pragma unroll
      for (int j = 0; j < 2; j++) {
        v16bf bv_ = load_frag(sVt + (wn + j * 16 + l15) * TAS + ks * 32 + hi * 8, 16);
        acc[j] = __builtin_amdgcn_wmma_f32_16x16x32_bf16(
            false, ap, false, bv_, (short)0, acc[j], false, false);
      }
    }
    __syncthreads();
  }

  // ---- normalize and store packed bf16 output ----
  #pragma unroll
  for (int j = 0; j < 2; j++)
    #pragma unroll
    for (int e = 0; e < 8; e++) {
      int row = m0 + e + (hi << 3);
      float v = acc[j][e] / sL[row];
      Op[(size_t)(b * SEQ + q0 + row) * DMODEL + h * DHEAD + wn + j * 16 + l15] = f2bf(v);
    }
}

// ---------------------------------------------------------------------------
// LayerNorm (+ optional residual): wave32 per row of 512.
// ---------------------------------------------------------------------------
template<bool HASRES, bool OUTB>
__global__ __launch_bounds__(256) void ln_resid(
    const float* __restrict__ y, const float* __restrict__ resid,
    const float* __restrict__ g, const float* __restrict__ bta,
    float* __restrict__ outf, unsigned short* __restrict__ outb, float eps)
{
  const int lane = threadIdx.x & 31;
  const int wave = threadIdx.x >> 5;
  const int row  = blockIdx.x * 8 + wave;
  const float* yr = y + (size_t)row * DMODEL;
  float vals[16];
  float s = 0.f, s2 = 0.f;
  #pragma unroll
  for (int i = 0; i < 16; i++) {
    float v = yr[lane + i * 32];
    vals[i] = v; s += v; s2 += v * v;
  }
  #pragma unroll
  for (int m = 16; m > 0; m >>= 1) {
    s  += __shfl_xor(s, m, 32);
    s2 += __shfl_xor(s2, m, 32);
  }
  float mu   = s  * (1.0f / DMODEL);
  float var  = s2 * (1.0f / DMODEL) - mu * mu;
  float rstd = rsqrtf(var + eps);
  #pragma unroll
  for (int i = 0; i < 16; i++) {
    int c = lane + i * 32;
    float v = (vals[i] - mu) * rstd * g[c] + bta[c];
    if (HASRES) v += resid[(size_t)row * DMODEL + c];
    outf[(size_t)row * DMODEL + c] = v;
    if (OUTB) outb[(size_t)row * DMODEL + c] = f2bf(v);
  }
}

// ---------------------------------------------------------------------------
// x = src + positional encoding; emits fp32 + bf16 copies.
// ---------------------------------------------------------------------------
__global__ void posenc_kernel(const float* __restrict__ src,
                              float* __restrict__ xf,
                              unsigned short* __restrict__ xb)
{
  int idx = blockIdx.x * blockDim.x + threadIdx.x;      // (b*S + s)*512 + c
  int c = idx & (DMODEL - 1);
  int s = (idx >> 9) & (SEQ - 1);
  float freq = __expf(-(float)(c & ~1) * (9.2103403720f / (float)DMODEL));
  float ang = (float)s * freq;
  float pe = (c & 1) ? __cosf(ang) : __sinf(ang);
  float v = src[idx] + pe;
  xf[idx] = v;
  xb[idx] = f2bf(v);
}

// ---------------------------------------------------------------------------
// Weight conversion (f32 -> bf16, transposed to [N][K])
// ---------------------------------------------------------------------------
// generic W [K,N] -> Wt [N,K];  idx = n*K + k, K = 1<<kshift
__global__ void cvt_w_t(const float* __restrict__ in, unsigned short* __restrict__ out,
                        int N, int kshift)
{
  int idx = blockIdx.x * blockDim.x + threadIdx.x;
  int n = idx >> kshift;
  int k = idx & ((1 << kshift) - 1);
  out[idx] = f2bf(in[(size_t)k * N + n]);
}

// QKV weight [H, D, DK] -> Wt [N=H*DK][K=D]
__global__ void cvt_qkv_weight_t(const float* __restrict__ in, unsigned short* __restrict__ out)
{
  int idx = blockIdx.x * blockDim.x + threadIdx.x;      // n*512 + d
  int n = idx >> 9;
  int d = idx & 511;
  int hh = n >> 6;
  int kk = n & 63;
  out[idx] = f2bf(in[(size_t)hh * DMODEL * DHEAD + d * DHEAD + kk]);
}

// ---------------------------------------------------------------------------
extern "C" void kernel_launch(void* const* d_in, const int* in_sizes, int n_in,
                              void* d_out, int out_size, void* d_ws, size_t ws_size,
                              hipStream_t stream)
{
  (void)in_sizes; (void)n_in; (void)out_size; (void)ws_size;
  const float* src  = (const float*)d_in[0];
  const float* Wq   = (const float*)d_in[1];
  const float* bq   = (const float*)d_in[2];
  const float* Wk   = (const float*)d_in[3];
  const float* bk   = (const float*)d_in[4];
  const float* Wv   = (const float*)d_in[5];
  const float* bv   = (const float*)d_in[6];
  const float* Wo   = (const float*)d_in[7];
  const float* bo   = (const float*)d_in[8];
  const float* ln1g = (const float*)d_in[9];
  const float* ln1b = (const float*)d_in[10];
  const float* W1   = (const float*)d_in[11];
  const float* b1   = (const float*)d_in[12];
  const float* W2   = (const float*)d_in[13];
  const float* b2   = (const float*)d_in[14];
  const float* ln2g = (const float*)d_in[15];
  const float* ln2b = (const float*)d_in[16];
  const float* lnfg = (const float*)d_in[17];
  const float* lnfb = (const float*)d_in[18];

  char* ws = (char*)d_ws;
  size_t off = 0;
  auto walloc = [&](size_t bytes) -> void* {
    void* p = ws + off;
    off += (bytes + 255) & ~(size_t)255;
    return p;
  };
  float*          xf  = (float*)walloc((size_t)BSROWS * DMODEL * 4);
  unsigned short* xb  = (unsigned short*)walloc((size_t)BSROWS * DMODEL * 2);
  unsigned short* qb  = (unsigned short*)walloc((size_t)BSROWS * DMODEL * 2);
  unsigned short* kb  = (unsigned short*)walloc((size_t)BSROWS * DMODEL * 2);
  unsigned short* vtb = (unsigned short*)walloc((size_t)BSROWS * DMODEL * 2);
  unsigned short* ob  = (unsigned short*)walloc((size_t)BSROWS * DMODEL * 2);
  float*          yf  = (float*)walloc((size_t)BSROWS * DMODEL * 4);
  unsigned short* hb  = (unsigned short*)walloc((size_t)BSROWS * DFFN * 2);
  unsigned short* wqb = (unsigned short*)walloc((size_t)DMODEL * DMODEL * 2);
  unsigned short* wkb = (unsigned short*)walloc((size_t)DMODEL * DMODEL * 2);
  unsigned short* wvb = (unsigned short*)walloc((size_t)DMODEL * DMODEL * 2);
  unsigned short* wob = (unsigned short*)walloc((size_t)DMODEL * DMODEL * 2);
  unsigned short* w1b = (unsigned short*)walloc((size_t)DMODEL * DFFN * 2);
  unsigned short* w2b = (unsigned short*)walloc((size_t)DFFN * DMODEL * 2);

  posenc_kernel<<<(BSROWS * DMODEL) / 256, 256, 0, stream>>>(src, xf, xb);

  const dim3 g512(DMODEL / GBN, BSROWS / GBM);
  const dim3 gffn(DFFN / GBN, BSROWS / GBM);
  const int nDD = DMODEL * DMODEL;        // 262144

  for (int l = 0; l < NLAYER; l++) {
    cvt_qkv_weight_t<<<nDD / 256, 256, 0, stream>>>(Wq + (size_t)l * nDD, wqb);
    cvt_qkv_weight_t<<<nDD / 256, 256, 0, stream>>>(Wk + (size_t)l * nDD, wkb);
    cvt_qkv_weight_t<<<nDD / 256, 256, 0, stream>>>(Wv + (size_t)l * nDD, wvb);
    cvt_w_t<<<nDD / 256, 256, 0, stream>>>(Wo + (size_t)l * nDD, wob, DMODEL, 9);
    cvt_w_t<<<(DMODEL * DFFN) / 256, 256, 0, stream>>>(W1 + (size_t)l * DMODEL * DFFN, w1b, DFFN, 9);
    cvt_w_t<<<(DFFN * DMODEL) / 256, 256, 0, stream>>>(W2 + (size_t)l * DFFN * DMODEL, w2b, DMODEL, 11);

    // Q/K projections packed [BS, H*DK] bf16; V written transposed [B,H,DK,S]
    gemm_bf16_wmma<true, false, false, true, false><<<g512, 256, 0, stream>>>(
        xb, wqb, bq + l * DMODEL, nullptr, qb, DMODEL, DMODEL);
    gemm_bf16_wmma<true, false, false, true, false><<<g512, 256, 0, stream>>>(
        xb, wkb, bk + l * DMODEL, nullptr, kb, DMODEL, DMODEL);
    gemm_bf16_wmma<true, false, false, false, true><<<g512, 256, 0, stream>>>(
        xb, wvb, bv + l * DMODEL, nullptr, vtb, DMODEL, DMODEL);

    // flash attention -> concat heads packed [BS, H*DK] bf16
    attn_wmma<<<dim3(SEQ / 64, NHEAD, NBATCH), 256, 0, stream>>>(qb, kb, vtb, ob, 0.125f);

    // output projection + LN1 residual
    gemm_bf16_wmma<true, false, true, false, false><<<g512, 256, 0, stream>>>(
        ob, wob, bo + l * DMODEL, yf, nullptr, DMODEL, DMODEL);
    ln_resid<true, true><<<BSROWS / 8, 256, 0, stream>>>(
        yf, xf, ln1g + l * DMODEL, ln1b + l * DMODEL, xf, xb, 1e-5f);

    // FFN
    gemm_bf16_wmma<true, true, false, true, false><<<gffn, 256, 0, stream>>>(
        xb, w1b, b1 + l * DFFN, nullptr, hb, DFFN, DMODEL);
    gemm_bf16_wmma<true, false, true, false, false><<<g512, 256, 0, stream>>>(
        hb, w2b, b2 + l * DMODEL, yf, nullptr, DMODEL, DFFN);
    ln_resid<true, true><<<BSROWS / 8, 256, 0, stream>>>(
        yf, xf, ln2g + l * DMODEL, ln2b + l * DMODEL, xf, xb, 1e-5f);
  }

  // final LayerNorm -> fp32 output
  ln_resid<false, false><<<BSROWS / 8, 256, 0, stream>>>(
      xf, nullptr, lnfg, lnfb, (float*)d_out, nullptr, 1e-6f);
}
